// GHOPVQVAEWrapper_67851893342898
// MI455X (gfx1250) — matrix operand order repaired and checked by reading, not compiled, not run on hardware
//
#include <hip/hip_runtime.h>
#include <hip/hip_bf16.h>
#include <cstdint>

// ---------------------------------------------------------------------------
// MI455X (gfx1250) VQ-VAE forward.
//   * conv3d = implicit GEMM on v_wmma_f32_16x16x32_f16 (wave32, 16x16 tiles)
//   * LDS tiles stored pre-swizzled in WMMA fragment order ->
//     per-lane fragment fetch is 2x ds_load_b128 (no scalar ds_load_u16 storm)
//   * branchless clamped im2col gathers (no per-element exec-mask branches)
//   * fused GroupNorm+SiLU, trilinear 2x upsample, VQ with LDS codebook tiles
// ---------------------------------------------------------------------------

typedef __attribute__((ext_vector_type(16))) _Float16 v16h;
typedef __attribute__((ext_vector_type(4)))  _Float16 v4h;
typedef __attribute__((ext_vector_type(8)))  float    v8f;

// =============================== conv3d =====================================
// GEMM view:  A = weights [Co x K], K = Ci*KSZ^3 (rows contiguous in OIDHW)
//             B = im2col  [K x N],  N = NB*Do*Ho*Wo
// Block: 128 threads = 4 waves. Block tile: 16 Co x 64 positions.
// LDS layout is fragment-order:
//   sAf[slot][e], slot = m + 16*(k>>4), e = k & 15
//   sBf[wave][slot][e], slot = n + 16*(k>>4), e = k & 15
// so lane `l` of wave `w` reads its v16h fragment contiguously.
template <int KSZ>
__global__ __launch_bounds__(128)
void conv3d_wmma_kernel(const float* __restrict__ in, const float* __restrict__ wgt,
                        const float* __restrict__ bias, const float* __restrict__ resid,
                        float* __restrict__ out,
                        int NB, int Ci, int Co, int Din, int Hin, int Win,
                        int Dout, int Hout, int Wout, int stride) {
  constexpr int K3 = KSZ * KSZ * KSZ;
  constexpr int K2 = KSZ * KSZ;
  constexpr int PAD = (KSZ - 1) / 2;
  const int K    = Ci * K3;
  const int Ntot = NB * Dout * Hout * Wout;
  const int HWin  = Hin * Win;
  const int HWout = Hout * Wout;

  const int tid  = threadIdx.x;
  const int wave = tid >> 5;
  const int lane = tid & 31;
  const int co0  = blockIdx.y << 4;
  const int nblk = blockIdx.x << 6;

  __shared__ __align__(16) _Float16 sAf[32 * 16];    // 1 KB
  __shared__ __align__(16) _Float16 sBf[128 * 16];   // 4 KB

  // ---- A staging geometry: thread -> slot = tid>>2, elements [aeq, aeq+3] ----
  const int aslot = tid >> 2;
  const int am    = aslot & 15;
  const int ahh   = aslot >> 4;
  const int aeq   = (tid & 3) << 2;                       // 0,4,8,12
  const int akoff = ahh * 16 + aeq;                       // 4 consecutive k
  const int arow  = co0 + am;
  const bool kmul4 = (K & 3) == 0;                        // float4 path legal

  // ---- B staging geometry: thread -> one column, 16 consecutive k ----
  const int bcol  = tid & 63;
  const int bhh   = tid >> 6;                             // k half (0: k 0..15, 1: 16..31)
  const int bslot = ((bcol >> 4) << 5) + (bcol & 15) + (bhh << 4);
  const int bpos  = nblk + bcol;
  const bool bvalid = bpos < Ntot;
  int b_ow, b_oh, b_od, b_b;
  {
    int tmp = bvalid ? bpos : 0;
    b_ow = tmp % Wout; tmp /= Wout;
    b_oh = tmp % Hout; tmp /= Hout;
    b_od = tmp % Dout; b_b = tmp / Dout;
  }
  const int base_id = b_od * stride - PAD;
  const int base_ih = b_oh * stride - PAD;
  const int base_iw = b_ow * stride - PAD;
  const size_t inb  = (size_t)b_b * Ci;

  v8f acc = {};

  for (int kk0 = 0; kk0 < K; kk0 += 32) {
    // ---------- stage A: 4 consecutive weights -> one ds_store_b64 ----------
    {
      const int k0 = kk0 + akoff;
      v4h hv;
      if (kmul4 && arow < Co && k0 + 3 < K) {
        const float4 w4 = *(const float4*)&wgt[(size_t)arow * K + k0];
        hv[0] = (_Float16)w4.x; hv[1] = (_Float16)w4.y;
        hv[2] = (_Float16)w4.z; hv[3] = (_Float16)w4.w;
      } else {
        #pragma unroll
        for (int j = 0; j < 4; ++j) {
          const int kk = k0 + j;
          const bool ok = (arow < Co) && (kk < K);
          const float v = wgt[ok ? ((size_t)arow * K + kk) : 0];
          hv[j] = (_Float16)(ok ? v : 0.f);
        }
      }
      *(v4h*)&sAf[aslot * 16 + aeq] = hv;
    }
    // ---------- stage B: 16 branchless gathers -> one 32B LDS store ----------
    {
      union { v16h v; _Float16 h[16]; } hb;
      #pragma unroll
      for (int j = 0; j < 16; ++j) {
        const int kk = kk0 + (bhh << 4) + j;
        int ci, kd, kh, kw;
        if (KSZ == 1) {
          ci = kk; kd = 0; kh = 0; kw = 0;
        } else {
          ci = kk / K3;                 // compile-time divisor -> magic multiply
          int t = kk - ci * K3;
          kd = t / K2; t -= kd * K2;
          kh = t / KSZ; kw = t - kh * KSZ;
        }
        const int id = base_id + kd;
        const int ih = base_ih + kh;
        const int iw = base_iw + kw;
        const bool ok = bvalid && (kk < K) &&
                        (unsigned)id < (unsigned)Din &&
                        (unsigned)ih < (unsigned)Hin &&
                        (unsigned)iw < (unsigned)Win;
        const size_t off = ok
            ? (((inb + ci) * Din + id) * (size_t)HWin + (size_t)ih * Win + iw)
            : 0;
        const float v = in[off];                    // unconditional, clamped
        hb.h[j] = (_Float16)(ok ? v : 0.f);
      }
      *(v16h*)&sBf[bslot * 16] = hb.v;
    }
    __syncthreads();

    // prefetch next weight chunk (global_prefetch_b8)
    if (kk0 + 32 < K)
      __builtin_prefetch(&wgt[(size_t)arow * K + kk0 + 32 + akoff], 0, 1);

    // ---------- fragments: contiguous 32B per lane -> ds_load_b128 x2 ----------
    const v16h fa = *(const v16h*)&sAf[lane * 16];
    const v16h fb = *(const v16h*)&sBf[(wave * 32 + lane) * 16];
    acc = __builtin_amdgcn_wmma_f32_16x16x32_f16(
        /*neg_a=*/false, fa, /*neg_b=*/false, fb,
        /*c_mod=*/(short)0, acc, /*reuse_a=*/false, /*reuse_b=*/false);
    __syncthreads();
  }

  // ---- store: C 16x16 f32 layout: lane = N col, VGPR r -> M = r + 8*(lane>>4)
  const int n   = lane & 15;
  const int hh  = lane >> 4;
  const int pos = nblk + (wave << 4) + n;
  if (pos < Ntot) {
    int ow = pos % Wout;
    int tmp = pos / Wout;
    int oh = tmp % Hout; tmp /= Hout;
    int od = tmp % Dout;
    int b  = tmp / Dout;
    #pragma unroll
    for (int r = 0; r < 8; ++r) {
      const int co = co0 + r + 8 * hh;
      if (co < Co) {
        const size_t off = (((size_t)b * Co + co) * Dout + od) * HWout
                         + (size_t)oh * Wout + ow;
        float v = acc[r] + bias[co];
        if (resid) v += resid[off];
        out[off] = v;
      }
    }
  }
}

// ========================= GroupNorm + SiLU (fused) =========================
__global__ __launch_bounds__(256)
void gn_silu_kernel(const float* __restrict__ in, const float* __restrict__ gamma,
                    const float* __restrict__ beta, float* __restrict__ out,
                    int C, int spatial, int groups, int cpg) {
  const int bg = blockIdx.x;
  const int b  = bg / groups;
  const int g  = bg % groups;
  const size_t base  = ((size_t)b * C + (size_t)g * cpg) * spatial;
  const size_t count = (size_t)cpg * spatial;

  __shared__ float rs[256], rq[256];
  float s = 0.f, q = 0.f;
  for (size_t i = threadIdx.x; i < count; i += 256) {
    float v = in[base + i];
    s += v; q += v * v;
  }
  rs[threadIdx.x] = s; rq[threadIdx.x] = q;
  __syncthreads();
  for (int off = 128; off > 0; off >>= 1) {
    if ((int)threadIdx.x < off) {
      rs[threadIdx.x] += rs[threadIdx.x + off];
      rq[threadIdx.x] += rq[threadIdx.x + off];
    }
    __syncthreads();
  }
  const float mean = rs[0] / (float)count;
  const float var  = rq[0] / (float)count - mean * mean;
  const float inv  = rsqrtf(var + 1e-5f);

  for (size_t i = threadIdx.x; i < count; i += 256) {
    const int c = g * cpg + (int)(i / (size_t)spatial);
    const float v = (in[base + i] - mean) * inv * gamma[c] + beta[c];
    out[base + i] = v / (1.f + __expf(-v));   // SiLU
  }
}

// ========================= trilinear 2x upsample ============================
__global__ __launch_bounds__(256)
void upsample2x_kernel(const float* __restrict__ in, float* __restrict__ out,
                       int NB, int C, int D, int H, int W) {
  const size_t total = (size_t)NB * C * 8u * D * H * W;
  size_t i = (size_t)blockIdx.x * blockDim.x + threadIdx.x;
  if (i >= total) return;
  const int W2 = 2 * W, H2 = 2 * H, D2 = 2 * D;
  int ow = (int)(i % W2); size_t t = i / W2;
  int oh = (int)(t % H2); t /= H2;
  int od = (int)(t % D2); t /= D2;
  int c  = (int)(t % C);
  int b  = (int)(t / C);

  // half-pixel centers: src = (dst + 0.5)/2 - 0.5
  float sd = od * 0.5f - 0.25f;
  float sh = oh * 0.5f - 0.25f;
  float sw = ow * 0.5f - 0.25f;
  int d0 = (int)floorf(sd); float fd = sd - (float)d0;
  int h0 = (int)floorf(sh); float fh = sh - (float)h0;
  int w0 = (int)floorf(sw); float fw = sw - (float)w0;
  int d1 = min(max(d0 + 1, 0), D - 1); d0 = min(max(d0, 0), D - 1);
  int h1 = min(max(h0 + 1, 0), H - 1); h0 = min(max(h0, 0), H - 1);
  int w1 = min(max(w0 + 1, 0), W - 1); w0 = min(max(w0, 0), W - 1);

  const size_t cb = ((size_t)b * C + c) * D;
  auto at = [&](int dd, int hh, int ww) -> float {
    return in[((cb + dd) * H + hh) * (size_t)W + ww];
  };
  float v =
      (1.f - fd) * ((1.f - fh) * ((1.f - fw) * at(d0, h0, w0) + fw * at(d0, h0, w1)) +
                    fh * ((1.f - fw) * at(d0, h1, w0) + fw * at(d0, h1, w1))) +
      fd * ((1.f - fh) * ((1.f - fw) * at(d1, h0, w0) + fw * at(d1, h0, w1)) +
            fh * ((1.f - fw) * at(d1, h1, w0) + fw * at(d1, h1, w1)));
  out[(((size_t)b * C + c) * D2 + od) * (size_t)H2 * W2 + (size_t)oh * W2 + ow] = v;
}

// ============================ vector quantize ===============================
// Forward: zq_st == zq; loss == 1.25 * MSE(zq, z) (both stop-grad terms equal).
__global__ __launch_bounds__(256)
void vq_kernel(const float* __restrict__ z, const float* __restrict__ cb,
               float* __restrict__ zq, int* __restrict__ idx_out,
               float* __restrict__ loss, int NB, int Dz, int Kc) {
  const int spatial = Dz * Dz * Dz;
  const int N = NB * spatial;
  const int p = blockIdx.x * blockDim.x + threadIdx.x;

  float z0 = 0.f, z1 = 0.f, z2 = 0.f;
  if (p < N) {
    int b = p / spatial, s = p - b * spatial;
    size_t base = (size_t)b * 3 * spatial + s;
    z0 = z[base];
    z1 = z[base + spatial];
    z2 = z[base + 2 * (size_t)spatial];
  }

  __shared__ float scb[2048 * 3];
  float best = 3.4e38f;
  int bidx = 0;
  for (int k0 = 0; k0 < Kc; k0 += 2048) {
    for (int i = threadIdx.x; i < 2048 * 3; i += 256)
      scb[i] = cb[(size_t)k0 * 3 + i];
    __syncthreads();
    for (int k = 0; k < 2048; ++k) {
      float d0 = z0 - scb[k * 3 + 0];
      float d1 = z1 - scb[k * 3 + 1];
      float d2 = z2 - scb[k * 3 + 2];
      float d = d0 * d0 + d1 * d1 + d2 * d2;
      if (d < best) { best = d; bidx = k0 + k; }   // strict < keeps first min
    }
    __syncthreads();
  }

  if (p < N) {
    int b = p / spatial, s = p - b * spatial;
    size_t base = (size_t)b * 3 * spatial + s;
    float c0 = cb[(size_t)bidx * 3 + 0];
    float c1 = cb[(size_t)bidx * 3 + 1];
    float c2 = cb[(size_t)bidx * 3 + 2];
    zq[base] = c0;
    zq[base + spatial] = c1;
    zq[base + 2 * (size_t)spatial] = c2;
    idx_out[p] = bidx;
    float l = (c0 - z0) * (c0 - z0) + (c1 - z1) * (c1 - z1) + (c2 - z2) * (c2 - z2);
    atomicAdd(loss, l * (1.25f / (float)(N * 3)));
  }
}

__global__ void init_loss_kernel(float* __restrict__ l) { *l = 0.f; }

// =============================== host side ==================================
extern "C" void kernel_launch(void* const* d_in, const int* in_sizes, int n_in,
                              void* d_out, int out_size, void* d_ws, size_t ws_size,
                              hipStream_t stream) {
  (void)in_sizes; (void)n_in; (void)out_size; (void)ws_size;
  const int NB = 4;

  auto P = [&](int i) -> const float* { return (const float*)d_in[i]; };
  auto cdiv = [](int a, int b) { return (a + b - 1) / b; };

  // ---- workspace carve (rotating activation buffers) ----
  float* ws = (float*)d_ws;
  const size_t SZ = 134217728ull;            // 128M floats (4 x 128ch x 64^3)
  float* A  = ws;
  float* Bb = ws + SZ;
  float* Cc = ws + 2 * SZ;
  float* Dd = ws + 3 * SZ;                   // skip-path buffer (<= 64M floats)
  float* zq = Dd + 67108864ull;              // 4 x 3 x 16^3 = 49152 floats

  float* outf = (float*)d_out;               // recon[1048576] | loss[1] | idx[16384]

  auto conv = [&](const float* in, const float* w, const float* bias, const float* res,
                  float* out, int Ci, int Co, int Din, int stride, int ksz) {
    const int pad  = (ksz - 1) / 2;
    const int Dout = (Din + 2 * pad - ksz) / stride + 1;
    const int N = NB * Dout * Dout * Dout;
    dim3 grid(cdiv(N, 64), cdiv(Co, 16));
    if (ksz == 3)
      conv3d_wmma_kernel<3><<<grid, 128, 0, stream>>>(in, w, bias, res, out,
          NB, Ci, Co, Din, Din, Din, Dout, Dout, Dout, stride);
    else
      conv3d_wmma_kernel<1><<<grid, 128, 0, stream>>>(in, w, bias, res, out,
          NB, Ci, Co, Din, Din, Din, Dout, Dout, Dout, stride);
  };
  auto gns = [&](const float* in, const float* g, const float* b, float* out,
                 int C, int Dsp) {
    const int spatial = Dsp * Dsp * Dsp;
    gn_silu_kernel<<<NB * 32, 256, 0, stream>>>(in, g, b, out, C, spatial, 32, C / 32);
  };
  auto ups = [&](const float* in, float* out, int C, int Dsp) {
    const size_t total = (size_t)NB * C * 8u * Dsp * Dsp * Dsp;
    upsample2x_kernel<<<(unsigned)((total + 255) / 256), 256, 0, stream>>>(
        in, out, NB, C, Dsp, Dsp, Dsp);
  };

  const float* x = P(0);

  // ---------------- encoder ----------------
  conv(x, P(1), P(2), nullptr, A, 16, 64, 64, 1, 3);              // A: 64ch 64^3
  // enc_r0 (64->64)
  gns(A, P(3), P(4), Bb, 64, 64);
  conv(Bb, P(5), P(6), nullptr, Cc, 64, 64, 64, 1, 3);
  gns(Cc, P(7), P(8), Bb, 64, 64);
  conv(Bb, P(9), P(10), A, Cc, 64, 64, 64, 1, 3);                 // Cc = out + x
  conv(Cc, P(11), P(12), nullptr, A, 64, 64, 64, 2, 3);           // A: 64ch 32^3
  // enc_r1 (64->128, skip conv)
  gns(A, P(13), P(14), Bb, 64, 32);
  conv(Bb, P(15), P(16), nullptr, Cc, 64, 128, 32, 1, 3);
  gns(Cc, P(17), P(18), Bb, 128, 32);
  conv(A, P(21), P(22), nullptr, Dd, 64, 128, 32, 1, 1);          // skip(x)
  conv(Bb, P(19), P(20), Dd, A, 128, 128, 32, 1, 3);              // A: 128ch 32^3
  conv(A, P(23), P(24), nullptr, Bb, 128, 128, 32, 2, 3);         // Bb: 128ch 16^3
  // enc_r2 (128->256, skip conv)
  gns(Bb, P(25), P(26), Cc, 128, 16);
  conv(Cc, P(27), P(28), nullptr, Dd, 128, 256, 16, 1, 3);
  gns(Dd, P(29), P(30), Cc, 256, 16);
  conv(Bb, P(33), P(34), nullptr, A, 128, 256, 16, 1, 1);         // skip(x)
  conv(Cc, P(31), P(32), A, Dd, 256, 256, 16, 1, 3);              // Dd: 256ch 16^3
  gns(Dd, P(35), P(36), A, 256, 16);
  conv(A, P(37), P(38), nullptr, Bb, 256, 3, 16, 1, 3);           // z = Bb [4,3,16^3]

  // ---------------- vector quantization ----------------
  init_loss_kernel<<<1, 1, 0, stream>>>(outf + 1048576);
  vq_kernel<<<64, 256, 0, stream>>>(Bb, P(39), zq,
                                    (int*)(outf + 1048577), outf + 1048576,
                                    NB, 16, 8192);

  // ---------------- decoder ----------------
  conv(zq, P(40), P(41), nullptr, A, 3, 256, 16, 1, 3);           // A: 256ch 16^3
  // dec_r2 (256->256)
  gns(A, P(42), P(43), Bb, 256, 16);
  conv(Bb, P(44), P(45), nullptr, Cc, 256, 256, 16, 1, 3);
  gns(Cc, P(46), P(47), Bb, 256, 16);
  conv(Bb, P(48), P(49), A, Cc, 256, 256, 16, 1, 3);              // Cc: 256ch 16^3
  // dec_u2
  ups(Cc, A, 256, 16);                                            // A: 256ch 32^3
  conv(A, P(50), P(51), nullptr, Bb, 256, 256, 32, 1, 3);         // Bb: 256ch 32^3
  // dec_r1 (256->128, skip conv)
  gns(Bb, P(52), P(53), Cc, 256, 32);
  conv(Cc, P(54), P(55), nullptr, A, 256, 128, 32, 1, 3);
  gns(A, P(56), P(57), Cc, 128, 32);
  conv(Bb, P(60), P(61), nullptr, Dd, 256, 128, 32, 1, 1);        // skip(x)
  conv(Cc, P(58), P(59), Dd, A, 128, 128, 32, 1, 3);              // A: 128ch 32^3
  // dec_u1
  ups(A, Bb, 128, 32);                                            // Bb: 128ch 64^3
  conv(Bb, P(62), P(63), nullptr, Cc, 128, 128, 64, 1, 3);        // Cc: 128ch 64^3
  // dec_r0 (128->64, skip conv)
  gns(Cc, P(64), P(65), A, 128, 64);
  conv(A, P(66), P(67), nullptr, Bb, 128, 64, 64, 1, 3);
  gns(Bb, P(68), P(69), A, 64, 64);
  conv(Cc, P(72), P(73), nullptr, Dd, 128, 64, 64, 1, 1);         // skip(x)
  conv(A, P(70), P(71), Dd, Bb, 64, 64, 64, 1, 3);                // Bb: 64ch 64^3
  gns(Bb, P(74), P(75), A, 64, 64);
  conv(A, P(76), P(77), nullptr, outf, 64, 1, 64, 1, 3);          // recon -> d_out
}
